// VisionTransformer_7799660610271
// MI455X (gfx1250) — matrix operand order repaired and checked by reading, not compiled, not run on hardware
//
#include <hip/hip_runtime.h>
#include <cstdint>
#include <cstddef>

// ---------------------------------------------------------------------------
// ViT forward (B=32, SEQ=197, HID=768, 12 heads, 12 blocks, MLP 3072, OUT 1000)
// f16 WMMA (v_wmma_f32_16x16x32_f16) for all GEMMs + fused attention.
// GEMM stages A/B tiles in LDS with double-buffered async-to-LDS copies
// (ASYNCcnt pipeline), fp32 residual / LN / softmax / GELU(erf) epilogues.
// ---------------------------------------------------------------------------

typedef _Float16 h16;
typedef __attribute__((ext_vector_type(16))) _Float16 v16h;
typedef __attribute__((ext_vector_type(8)))  _Float16 v8h;
typedef __attribute__((ext_vector_type(8)))  float    v8f;

#define HIDDEN   768
#define HEADS    12
#define DKH      64
#define NBLOCKS  12
#define SEQLEN   197
#define SEQPAD   224            // 7 * 32  (K-dim padding for P@V WMMA)
#define BATCH    32
#define NPATCH   196
#define MTOK     6304           // 32 * 197  (multiple of 16)
#define MROWS    6336           // MTOK padded to multiple of 64
#define MPAT     6272           // 32 * 196  (multiple of 64)
#define MLPD     3072
#define OUTN     1000
#define OUTPAD   1024           // head weight rows padded to 8*128

// ------------------------- CDNA5 async-to-LDS path -------------------------

#if __has_builtin(__builtin_amdgcn_s_wait_asynccnt)
#define WAIT_ASYNC0() __builtin_amdgcn_s_wait_asynccnt(0)
#else
#define WAIT_ASYNC0() asm volatile("s_wait_asynccnt 0x0" ::: "memory")
#endif

// One 16-byte async copy global -> LDS (GLOBAL_LOAD_ASYNC_TO_LDS_B128).
// VDST VGPR holds the raw LDS byte offset (= low 32 bits of the generic
// shared-space pointer; flat LDS aperture keeps the offset in addr[31:0]).
__device__ __forceinline__ void async_load_b128(const h16* g, h16* l) {
  uint32_t loff = (uint32_t)(uintptr_t)l;
  asm volatile("global_load_async_to_lds_b128 %0, %1, off"
               :: "v"(loff), "v"((uint64_t)(uintptr_t)g)
               : "memory");
}

// ------------------------------- WMMA helpers ------------------------------

__device__ __forceinline__ v8f zero8() {
  v8f z;
#pragma unroll
  for (int i = 0; i < 8; ++i) z[i] = 0.0f;
  return z;
}

__device__ __forceinline__ v8f wmma_f16(v16h a, v16h b, v8f c) {
  // D = A(16x32, f16) x B(32x16, f16) + C(16x16, f32)
  return __builtin_amdgcn_wmma_f32_16x16x32_f16(
      /*neg_a=*/false, a, /*neg_b=*/false, b,
      /*c_mod=*/(short)0, c, /*reuse_a=*/false, /*reuse_b=*/false);
}

// A fragment: 16x32 f16 tile, row-major source with leading dim lda (halfs).
// Contiguous-K layout used here: lane<16 -> row=lane, K = kb+0..15
//             lane>=16 -> row=lane-16, K = kb+16..31
__device__ __forceinline__ v16h load_a_frag(const h16* base, int lda, int kb, int lane) {
  int m  = lane & 15;
  int hi = lane >> 4;
  return *(const v16h*)(base + (size_t)m * lda + kb + hi * 16);
}

// B fragment: 32x16, where B[k][n] = W[n][k], W row-major with leading dim ldw.
// lane<16 -> col n = n0+lane,     K = kb+0..15  (contiguous 32B of row n)
// lane>=16-> col n = n0+lane-16,  K = kb+16..31
__device__ __forceinline__ v16h load_b_frag(const h16* W, int ldw, int n0, int kb,
                                            int lane) {
  int n  = n0 + (lane & 15);
  int hi = lane >> 4;
  return *(const v16h*)(W + (size_t)n * ldw + kb + hi * 16);
}

// ------------------------------- GEMM kernel -------------------------------
// Out = act(A @ W^T + bias [+ resid]); A [M,K] f16, W [N,K] f16.
// Block: 256 threads = 8 waves (4 M-waves x 2 N-waves), wave tile 16x64,
// block tile 64x128. K-step 32. A (64x32) and B (128x32) tiles staged in LDS
// via double-buffered async-to-LDS copies; fragments read from LDS.
// All fragment loads are unguarded: callers pad M to 64, N to 128, K to 32.

__device__ __forceinline__ void stage_tiles(const h16* A, const h16* W, int K,
                                            int rowBlock, int colBlock, int kb,
                                            h16* sa, h16* sb, int tid) {
  {
    int row = tid >> 2, q = tid & 3;                    // 64 rows x 4 chunks
    async_load_b128(A + (size_t)(rowBlock + row) * K + kb + q * 8,
                    sa + row * 32 + q * 8);
  }
#pragma unroll
  for (int t = 0; t < 2; ++t) {                         // 128 rows x 4 chunks
    int chunk = tid + t * 256;
    int row = chunk >> 2, q = chunk & 3;
    async_load_b128(W + (size_t)(colBlock + row) * K + kb + q * 8,
                    sb + row * 32 + q * 8);
  }
}

__global__ __launch_bounds__(256) void gemm_kernel(
    const h16* __restrict__ A, const h16* __restrict__ W,
    const float* __restrict__ bias, const float* __restrict__ resid,
    float* __restrict__ outF, h16* __restrict__ outH,
    int M, int N, int K, int doGelu) {
  __shared__ __align__(32) h16 sA[2][64 * 32];    // 4 KB per buffer
  __shared__ __align__(32) h16 sB[2][128 * 32];   // 8 KB per buffer

  int tid  = threadIdx.x;
  int lane = tid & 31;
  int wave = tid >> 5;
  int mw = wave & 3;
  int nw = wave >> 2;
  int rowBlock = blockIdx.y * 64;
  int colBlock = blockIdx.x * 128;

  v8f acc[4];
#pragma unroll
  for (int j = 0; j < 4; ++j) acc[j] = zero8();

  stage_tiles(A, W, K, rowBlock, colBlock, 0, sA[0], sB[0], tid);

  int nsteps = K >> 5;
  for (int s = 0; s < nsteps; ++s) {
    WAIT_ASYNC0();        // own async copies for step s complete
    __syncthreads();      // tiles visible; everyone done reading other buffer
    if (s + 1 < nsteps)
      stage_tiles(A, W, K, rowBlock, colBlock, (s + 1) << 5,
                  sA[(s + 1) & 1], sB[(s + 1) & 1], tid);

    const h16* sa = sA[s & 1] + mw * 16 * 32;
    const h16* sb = sB[s & 1] + nw * 64 * 32;
    v16h a = load_a_frag(sa, 32, 0, lane);
#pragma unroll
    for (int j = 0; j < 4; ++j) {
      v16h b = load_b_frag(sb, 32, j * 16, 0, lane);
      acc[j] = wmma_f16(a, b, acc[j]);
    }
  }

  int hi = lane >> 4;
  int cl = lane & 15;
  int rowBase = rowBlock + mw * 16;
#pragma unroll
  for (int j = 0; j < 4; ++j) {
    int col = colBlock + nw * 64 + j * 16 + cl;
#pragma unroll
    for (int i = 0; i < 8; ++i) {
      int row = rowBase + i + hi * 8;
      if (row < M && col < N) {
        float v = acc[j][i];
        if (bias) v += bias[col];
        if (doGelu) v = 0.5f * v * (1.0f + erff(v * 0.70710678118654752f));
        size_t idx = (size_t)row * N + col;
        if (resid) v += resid[idx];
        if (outF) outF[idx] = v;
        if (outH) outH[idx] = (h16)v;
      }
    }
  }
}

// --------------------------- fused attention kernel ------------------------
// grid = (13 m-tiles, HEADS, BATCH), block = 32 (one wave).
// scores(16xSEQPAD) -> LDS -> softmax(fp32, 197 valid cols) -> P f16 in LDS
// -> P @ V  via WMMA against pre-transposed zero-padded Vt [64 x SEQPAD].

__global__ __launch_bounds__(32) void attn_kernel(
    const h16* __restrict__ Q, const h16* __restrict__ Km,
    const h16* __restrict__ Vt, h16* __restrict__ Out) {
  __shared__ __align__(32) float sS[16 * SEQPAD];
  __shared__ __align__(32) h16   sP[16 * SEQPAD];

  int lane = threadIdx.x & 31;
  int mt = blockIdx.x;
  int h  = blockIdx.y;
  int b  = blockIdx.z;

  const h16* Qb = Q  + ((size_t)(b * SEQLEN + mt * 16)) * HIDDEN + h * DKH;
  const h16* Kb = Km + ((size_t)(b * SEQLEN)) * HIDDEN + h * DKH;
  const h16* Vb = Vt + ((size_t)(b * HEADS + h)) * DKH * SEQPAD;

  v16h a0 = load_a_frag(Qb, HIDDEN, 0, lane);
  v16h a1 = load_a_frag(Qb, HIDDEN, 32, lane);

  const float scale = 0.125f;  // 1/sqrt(64)
  int hi = lane >> 4;
  int cl = lane & 15;

  // scores = Q @ K^T (cols >= 197 are garbage, never read by softmax)
  for (int nt = 0; nt < SEQPAD / 16; ++nt) {
    v8f acc = zero8();
    v16h b0 = load_b_frag(Kb, HIDDEN, nt * 16, 0, lane);
    acc = wmma_f16(a0, b0, acc);
    v16h b1 = load_b_frag(Kb, HIDDEN, nt * 16, 32, lane);
    acc = wmma_f16(a1, b1, acc);
#pragma unroll
    for (int i = 0; i < 8; ++i)
      sS[(i + hi * 8) * SEQPAD + nt * 16 + cl] = acc[i] * scale;
  }

  // softmax per query row (single wave: LDS ops are in-order per wave)
  if (lane < 16) {
    int q = mt * 16 + lane;
    float* sr = sS + lane * SEQPAD;
    h16*   pr = sP + lane * SEQPAD;
    if (q < SEQLEN) {
      float mx = -1e30f;
      for (int c = 0; c < SEQLEN; ++c) mx = fmaxf(mx, sr[c]);
      float sum = 0.0f;
      for (int c = 0; c < SEQLEN; ++c) {
        float e = __expf(sr[c] - mx);
        sr[c] = e;
        sum += e;
      }
      float inv = 1.0f / sum;
      for (int c = 0; c < SEQPAD; ++c)
        pr[c] = (c < SEQLEN) ? (h16)(sr[c] * inv) : (h16)0.0f;
    } else {
      for (int c = 0; c < SEQPAD; ++c) pr[c] = (h16)0.0f;
    }
  }

  // out(16 x 64) = P(16 x SEQPAD) @ V(SEQPAD x 64); B frag from Vt rows.
  v8f o[4];
#pragma unroll
  for (int j = 0; j < 4; ++j) o[j] = zero8();
  for (int kc = 0; kc < SEQPAD; kc += 32) {
    v16h a = load_a_frag(sP, SEQPAD, kc, lane);
#pragma unroll
    for (int j = 0; j < 4; ++j) {
      v16h bb = load_b_frag(Vb, SEQPAD, j * 16, kc, lane);
      o[j] = wmma_f16(a, bb, o[j]);
    }
  }

#pragma unroll
  for (int j = 0; j < 4; ++j) {
#pragma unroll
    for (int i = 0; i < 8; ++i) {
      int row = mt * 16 + i + hi * 8;
      if (row < SEQLEN)
        Out[((size_t)(b * SEQLEN + row)) * HIDDEN + h * DKH + j * 16 + cl] =
            (h16)o[j][i];
    }
  }
}

// ------------------------------ small kernels ------------------------------

__global__ void cvt_kernel(const float* __restrict__ src, h16* __restrict__ dst, int n) {
  int i = blockIdx.x * blockDim.x + threadIdx.x;
  if (i < n) dst[i] = (h16)src[i];
}

// convert with zero padding (used for the classifier head: 1000 -> 1024 rows)
__global__ void cvt_pad_kernel(const float* __restrict__ src, h16* __restrict__ dst,
                               int n, int npad) {
  int i = blockIdx.x * blockDim.x + threadIdx.x;
  if (i < npad) dst[i] = (i < n) ? (h16)src[i] : (h16)0.0f;
}

__global__ void patchify_kernel(const float* __restrict__ img, h16* __restrict__ dst) {
  int idx = blockIdx.x * blockDim.x + threadIdx.x;
  if (idx >= BATCH * NPATCH * HIDDEN) return;
  int col = idx % HIDDEN;
  int row = idx / HIDDEN;
  int b = row / NPATCH;
  int pidx = row % NPATCH;
  int c = col >> 8;           // 256 elems per channel
  int rem = col & 255;
  int iy = rem >> 4;
  int ix = rem & 15;
  int py = pidx / 14;
  int px = pidx % 14;
  dst[idx] = (h16)img[(((size_t)b * 3 + c) * 224 + (py * 16 + iy)) * 224 + (px * 16 + ix)];
}

__global__ void assemble_kernel(const float* __restrict__ tokens,
                                const float* __restrict__ cls,
                                float* __restrict__ x) {
  int idx = blockIdx.x * blockDim.x + threadIdx.x;
  if (idx >= MTOK * HIDDEN) return;
  int col = idx % HIDDEN;
  int row = idx / HIDDEN;
  int b = row / SEQLEN;
  int n = row % SEQLEN;
  // sin/cos positional encoding
  float div = __expf((float)(col & ~1) * (-9.210340371976184f / 768.0f));
  float ang = (float)n * div;
  float pe = (col & 1) ? __cosf(ang) : __sinf(ang);
  float base = (n == 0) ? cls[col]
                        : tokens[((size_t)(b * NPATCH + n - 1)) * HIDDEN + col];
  x[idx] = base + pe;
}

__global__ __launch_bounds__(256) void ln_kernel(
    const float* __restrict__ x, const float* __restrict__ g,
    const float* __restrict__ bta, h16* __restrict__ yh) {
  __shared__ float red1[8];
  __shared__ float red2[8];
  int row = blockIdx.x;
  const float* xr = x + (size_t)row * HIDDEN;
  float s = 0.0f, s2 = 0.0f;
  for (int c = threadIdx.x; c < HIDDEN; c += 256) {
    float v = xr[c];
    s += v;
    s2 += v * v;
  }
  for (int o = 16; o > 0; o >>= 1) {
    s  += __shfl_down(s, o, 32);
    s2 += __shfl_down(s2, o, 32);
  }
  int lane = threadIdx.x & 31, wv = threadIdx.x >> 5;
  if (lane == 0) { red1[wv] = s; red2[wv] = s2; }
  __syncthreads();
  if (wv == 0) {
    s  = (lane < 8) ? red1[lane] : 0.0f;
    s2 = (lane < 8) ? red2[lane] : 0.0f;
    for (int o = 4; o > 0; o >>= 1) {
      s  += __shfl_down(s, o, 32);
      s2 += __shfl_down(s2, o, 32);
    }
    if (lane == 0) { red1[0] = s; red2[0] = s2; }
  }
  __syncthreads();
  float mean = red1[0] * (1.0f / HIDDEN);
  float var  = red2[0] * (1.0f / HIDDEN) - mean * mean;
  float inv  = rsqrtf(var + 1e-5f);
  for (int c = threadIdx.x; c < HIDDEN; c += 256)
    yh[(size_t)row * HIDDEN + c] = (h16)(g[c] * ((xr[c] - mean) * inv) + bta[c]);
}

// pack V [MTOK, HIDDEN] f16 -> Vt [B, H, DKH, SEQPAD] f16, zero pad cols>=197
__global__ void vpack_kernel(const h16* __restrict__ vh, h16* __restrict__ vt) {
  int idx = blockIdx.x * blockDim.x + threadIdx.x;
  if (idx >= BATCH * HEADS * DKH * SEQPAD) return;
  int n = idx % SEQPAD;
  int d = (idx / SEQPAD) % DKH;
  int h = (idx / (SEQPAD * DKH)) % HEADS;
  int b = idx / (SEQPAD * DKH * HEADS);
  h16 v = (h16)0.0f;
  if (n < SEQLEN) v = vh[((size_t)(b * SEQLEN + n)) * HIDDEN + h * DKH + d];
  vt[idx] = v;
}

// extract CLS rows (f32 -> f16), zero-pad to 64 rows for fragment loads
__global__ void clsext_kernel(const float* __restrict__ x, h16* __restrict__ cls) {
  int idx = blockIdx.x * blockDim.x + threadIdx.x;
  if (idx >= 64 * HIDDEN) return;
  int r = idx / HIDDEN;
  int c = idx % HIDDEN;
  cls[idx] = (r < BATCH) ? (h16)x[((size_t)r * SEQLEN) * HIDDEN + c] : (h16)0.0f;
}

__global__ __launch_bounds__(256) void softmax_kernel(
    const float* __restrict__ logits, float* __restrict__ out) {
  __shared__ float red[8];
  __shared__ float bc;
  int row = blockIdx.x;
  const float* lr = logits + (size_t)row * OUTN;
  int lane = threadIdx.x & 31, wv = threadIdx.x >> 5;

  float mx = -1e30f;
  for (int c = threadIdx.x; c < OUTN; c += 256) mx = fmaxf(mx, lr[c]);
  for (int o = 16; o > 0; o >>= 1) mx = fmaxf(mx, __shfl_down(mx, o, 32));
  if (lane == 0) red[wv] = mx;
  __syncthreads();
  if (threadIdx.x == 0) {
    float m = red[0];
    for (int i = 1; i < 8; ++i) m = fmaxf(m, red[i]);
    bc = m;
  }
  __syncthreads();
  mx = bc;

  float sum = 0.0f;
  for (int c = threadIdx.x; c < OUTN; c += 256) sum += __expf(lr[c] - mx);
  for (int o = 16; o > 0; o >>= 1) sum += __shfl_down(sum, o, 32);
  if (lane == 0) red[wv] = sum;
  __syncthreads();
  if (threadIdx.x == 0) {
    float s = 0.0f;
    for (int i = 0; i < 8; ++i) s += red[i];
    bc = s;
  }
  __syncthreads();
  float inv = 1.0f / bc;
  for (int c = threadIdx.x; c < OUTN; c += 256)
    out[(size_t)row * OUTN + c] = __expf(lr[c] - mx) * inv;
}

// -------------------------------- host side --------------------------------

static inline void launch_gemm(const h16* A, const h16* W, const float* bias,
                               const float* resid, float* outF, h16* outH,
                               int M, int N, int K, int gelu, hipStream_t st) {
  dim3 grid((N + 127) / 128, (M + 63) / 64);
  gemm_kernel<<<grid, dim3(256), 0, st>>>(A, W, bias, resid, outF, outH, M, N, K, gelu);
}

static inline void launch_cvt(const float* s, h16* d, int n, hipStream_t st) {
  cvt_kernel<<<(n + 255) / 256, 256, 0, st>>>(s, d, n);
}

extern "C" void kernel_launch(void* const* d_in, const int* in_sizes, int n_in,
                              void* d_out, int out_size, void* d_ws, size_t ws_size,
                              hipStream_t stream) {
  (void)in_sizes; (void)n_in; (void)out_size; (void)ws_size;

  const float* images    = (const float*)d_in[0];
  const float* W_map     = (const float*)d_in[1];
  const float* b_map     = (const float*)d_in[2];
  const float* cls_token = (const float*)d_in[3];
  const float* ln1_g     = (const float*)d_in[4];
  const float* ln1_b     = (const float*)d_in[5];
  const float* Wq        = (const float*)d_in[6];
  const float* Wk        = (const float*)d_in[7];
  const float* Wv        = (const float*)d_in[8];
  const float* Wo_w      = (const float*)d_in[9];
  const float* Wo_b      = (const float*)d_in[10];
  const float* ln2_g     = (const float*)d_in[11];
  const float* ln2_b     = (const float*)d_in[12];
  const float* W1        = (const float*)d_in[13];
  const float* b1        = (const float*)d_in[14];
  const float* W2        = (const float*)d_in[15];
  const float* b2        = (const float*)d_in[16];
  const float* head_w    = (const float*)d_in[17];
  const float* head_b    = (const float*)d_in[18];
  float* out = (float*)d_out;

  char* ws = (char*)d_ws;
  size_t off = 0;
  auto alloc = [&](size_t bytes) -> void* {
    void* p = (void*)(ws + off);
    off += (bytes + 255) & ~(size_t)255;
    return p;
  };

  h16*   wmap_h  = (h16*)alloc((size_t)HIDDEN * HIDDEN * 2);
  h16*   patch_h = (h16*)alloc((size_t)MPAT * HIDDEN * 2);
  float* tokensf = (float*)alloc((size_t)MPAT * HIDDEN * 4);
  float* xf      = (float*)alloc((size_t)MROWS * HIDDEN * 4);
  h16*   yh      = (h16*)alloc((size_t)MROWS * HIDDEN * 2);
  h16*   qh      = (h16*)alloc((size_t)MROWS * HIDDEN * 2);
  h16*   kh      = (h16*)alloc((size_t)MROWS * HIDDEN * 2);
  h16*   vh      = (h16*)alloc((size_t)MROWS * HIDDEN * 2);
  h16*   vt      = (h16*)alloc((size_t)BATCH * HEADS * DKH * SEQPAD * 2);
  h16*   attn_h  = (h16*)alloc((size_t)MROWS * HIDDEN * 2);
  h16*   mlp_h   = (h16*)alloc((size_t)MROWS * MLPD * 2);
  h16*   wq_h    = (h16*)alloc((size_t)HIDDEN * HIDDEN * 2);
  h16*   wk_h    = (h16*)alloc((size_t)HIDDEN * HIDDEN * 2);
  h16*   wv_h    = (h16*)alloc((size_t)HIDDEN * HIDDEN * 2);
  h16*   wo_h    = (h16*)alloc((size_t)HIDDEN * HIDDEN * 2);
  h16*   w1_h    = (h16*)alloc((size_t)MLPD * HIDDEN * 2);
  h16*   w2_h    = (h16*)alloc((size_t)MLPD * HIDDEN * 2);
  h16*   cls_h   = (h16*)alloc((size_t)64 * HIDDEN * 2);
  h16*   headw_h = (h16*)alloc((size_t)OUTPAD * HIDDEN * 2);
  float* logits  = (float*)alloc((size_t)BATCH * OUTN * 4);

  // ---- patch embedding: tokens = patches @ W_map^T + b_map ----
  launch_cvt(W_map, wmap_h, HIDDEN * HIDDEN, stream);
  {
    int n = BATCH * NPATCH * HIDDEN;
    patchify_kernel<<<(n + 255) / 256, 256, 0, stream>>>(images, patch_h);
  }
  launch_gemm(patch_h, wmap_h, b_map, nullptr, tokensf, nullptr,
              MPAT, HIDDEN, HIDDEN, 0, stream);
  {
    int n = MTOK * HIDDEN;
    assemble_kernel<<<(n + 255) / 256, 256, 0, stream>>>(tokensf, cls_token, xf);
  }

  // ---- transformer blocks ----
  for (int i = 0; i < NBLOCKS; ++i) {
    const size_t wsz  = (size_t)HIDDEN * HIDDEN;   // 589824
    const size_t msz  = (size_t)MLPD * HIDDEN;     // 2359296
    launch_cvt(Wq + (size_t)i * wsz, wq_h, (int)wsz, stream);
    launch_cvt(Wk + (size_t)i * wsz, wk_h, (int)wsz, stream);
    launch_cvt(Wv + (size_t)i * wsz, wv_h, (int)wsz, stream);
    launch_cvt(Wo_w + (size_t)i * wsz, wo_h, (int)wsz, stream);
    launch_cvt(W1 + (size_t)i * msz, w1_h, (int)msz, stream);
    launch_cvt(W2 + (size_t)i * msz, w2_h, (int)msz, stream);

    // y = LN1(x)
    ln_kernel<<<MTOK, 256, 0, stream>>>(xf, ln1_g + i * HIDDEN, ln1_b + i * HIDDEN, yh);

    // Q, K, V projections (no bias in reference)
    launch_gemm(yh, wq_h, nullptr, nullptr, nullptr, qh, MTOK, HIDDEN, HIDDEN, 0, stream);
    launch_gemm(yh, wk_h, nullptr, nullptr, nullptr, kh, MTOK, HIDDEN, HIDDEN, 0, stream);
    launch_gemm(yh, wv_h, nullptr, nullptr, nullptr, vh, MTOK, HIDDEN, HIDDEN, 0, stream);

    {
      int n = BATCH * HEADS * DKH * SEQPAD;
      vpack_kernel<<<(n + 255) / 256, 256, 0, stream>>>(vh, vt);
    }

    // fused softmax(QK^T/sqrt(dk)) @ V
    attn_kernel<<<dim3((SEQLEN + 15) / 16, HEADS, BATCH), dim3(32), 0, stream>>>(
        qh, kh, vt, attn_h);

    // x = x + attn @ Wo^T + Wo_b
    launch_gemm(attn_h, wo_h, Wo_b + i * HIDDEN, xf, xf, nullptr,
                MTOK, HIDDEN, HIDDEN, 0, stream);

    // y = LN2(x)
    ln_kernel<<<MTOK, 256, 0, stream>>>(xf, ln2_g + i * HIDDEN, ln2_b + i * HIDDEN, yh);

    // h = gelu(y @ W1^T + b1) ; x = x + h @ W2^T + b2
    launch_gemm(yh, w1_h, b1 + (size_t)i * MLPD, nullptr, nullptr, mlp_h,
                MTOK, MLPD, HIDDEN, 1, stream);
    launch_gemm(mlp_h, w2_h, b2 + i * HIDDEN, xf, xf, nullptr,
                MTOK, HIDDEN, MLPD, 0, stream);
  }

  // ---- classifier head + softmax ----
  {
    int n = 64 * HIDDEN;
    clsext_kernel<<<(n + 255) / 256, 256, 0, stream>>>(xf, cls_h);
  }
  {
    int n = OUTPAD * HIDDEN;
    cvt_pad_kernel<<<(n + 255) / 256, 256, 0, stream>>>(head_w, headw_h,
                                                        OUTN * HIDDEN, n);
  }
  launch_gemm(cls_h, headw_h, head_b, nullptr, logits, nullptr,
              BATCH, OUTN, HIDDEN, 0, stream);
  softmax_kernel<<<BATCH, 256, 0, stream>>>(logits, out);
}
